// NCCLoss_22857815949864
// MI455X (gfx1250) — compile-verified, matrix operand order checked by reading, not run on hardware
//
#include <hip/hip_runtime.h>
#include <stdint.h>

// ---------------- problem geometry ----------------
#define WDIM 192
#define HDIM 192
#define DDIM 192
#define BATCH 2
#define TS 16          // output tile side (H,W)
#define HALO 4         // 9x9x9 box -> +-4
#define TIN 24         // TS + 2*HALO
#define PW 16          // wsum row pitch (floats)
#define NT 12          // 192/16 tiles per axis
#define DCH 48         // depth chunk per workgroup
#define NDC 4          // 192/48
#define NBLK (NT * NT * NDC * BATCH)   // 1152 workgroups
#define KSUMF 729.0f
#define NTOT 14155776.0                // 2*192^3

typedef float    v2f  __attribute__((ext_vector_type(2)));
typedef float    v8f  __attribute__((ext_vector_type(8)));
typedef _Float16 v16h __attribute__((ext_vector_type(16)));
typedef unsigned int u32x4 __attribute__((ext_vector_type(4)));
typedef int      i32x8 __attribute__((ext_vector_type(8)));
typedef int      i32x4 __attribute__((ext_vector_type(4)));

#if defined(__has_builtin)
#if __has_builtin(__builtin_amdgcn_wmma_f32_16x16x4_f32)
#define NCC_WMMA_F32X4 1
#endif
#endif

// LDS byte offset of a shared-memory object (generic -> addrspace(3) -> int).
__device__ __forceinline__ uint32_t lds_addr_of(void* p) {
  return (uint32_t)(uintptr_t)(__attribute__((address_space(3))) char*)p;
}

// TDM: async DMA of a 2D tile (cw x ch f32, global row stride = WDIM) into LDS,
// rows packed contiguously (pitch = cw). Tracked by TENSORcnt.
__device__ __forceinline__ void tdm_load_tile_2d(const float* gsrc, uint32_t lds_byte,
                                                 int cw, int ch) {
  uint64_t ga = (uint64_t)(uintptr_t)gsrc;
  u32x4 g0;
  g0[0] = 1u;                                                   // count=1 (valid user D#)
  g0[1] = lds_byte;                                             // lds_addr
  g0[2] = (uint32_t)ga;                                         // global_addr[31:0]
  g0[3] = (uint32_t)((ga >> 32) & 0x01FFFFFFull) | (2u << 30);  // ga[56:32], type=2
  i32x8 g1;
  g1[0] = (2 << 16);          // workgroup_mask=0, data_size=2 (4B), no iterate/pad
  g1[1] = 0;                  // tensor_dim0 = 1<<20 : [15:0] here
  g1[2] = 16;                 // tensor_dim0[31:16]=16, tensor_dim1[15:0]=0
  g1[3] = 16 | (cw << 16);    // tensor_dim1[31:16]=16 (1<<20), tile_dim0=cw
  g1[4] = ch;                 // tile_dim1=ch, tile_dim2=0
  g1[5] = WDIM;               // tensor_dim0_stride (elements), lo32
  g1[6] = 0;
  g1[7] = 0;
  i32x4 g2 = {0, 0, 0, 0};
  i32x4 g3 = {0, 0, 0, 0};
  i32x8 g4 = {0, 0, 0, 0, 0, 0, 0, 0};   // extended group (unused / non-cluster)
  __builtin_amdgcn_tensor_load_to_lds(g0, g1, g2, g3, g4, 0);
}

// One wave32 per workgroup. Fused: TDM slab load -> pointwise products ->
// separable 9-tap box sums (W: VALU sliding window, H: WMMA banded matmul,
// D: LDS ring + register running sums) -> NCC pointwise -> partial sum.
__global__ __launch_bounds__(32) void ncc_main(const float* __restrict__ gI,
                                               const float* __restrict__ gJ,
                                               float* __restrict__ partials) {
  __shared__ float sI[2][TIN * TIN];      // double-buffered input slabs (pitch = cw)
  __shared__ float sJ[2][TIN * TIN];
  __shared__ float sW[5][TIN * PW];       // W-axis sums, 24 rows x 16 cols per quantity
  __shared__ float sRing[9][5][8 * 32];   // box-2D slabs, register-layout (lane*8+k)
  __shared__ float sCenI[5][8 * 32];      // center-value lag ring (lag 4)
  __shared__ float sCenJ[5][8 * 32];

  const int lane = threadIdx.x;
  const int hi = lane >> 4;
  const int lo = lane & 15;

  const int bid = blockIdx.x;
  const int tw = bid % NT;
  const int th = (bid / NT) % NT;
  const int dz = (bid / (NT * NT)) % NDC;
  const int b  = bid / (NT * NT * NDC);

  const int h0 = th * TS, w0 = tw * TS;
  const int r0 = max(h0 - HALO, 0);
  const int r1 = min(h0 + TS + HALO, HDIM);
  const int c0 = max(w0 - HALO, 0);
  const int c1 = min(w0 + TS + HALO, WDIM);
  const int chh = r1 - r0, cww = c1 - c0;  // clamped tile extents (<= 24)
  const int lro = r0 - (h0 - HALO);        // rows clipped at top (0 or 4)
  const int lco = c0 - (w0 - HALO);        // cols clipped at left (0 or 4)

  // zero the ring so the first 9 running-sum subtractions are no-ops
  for (int i = lane; i < 9 * 5 * 256; i += 32) (&sRing[0][0][0])[i] = 0.0f;

  // ---- banded-ones matrix for the H-axis matmul: Band[m][k]=1 iff m<=k<=m+8 ----
#ifdef NCC_WMMA_F32X4
  v2f bandA[6];   // A operand, 16x4 f32 slices: lane L -> M=L&15, K=4s+2*(L>>4)+{0,1}
#pragma unroll
  for (int s = 0; s < 6; ++s) {
    int k0 = 4 * s + 2 * hi;
    bandA[s].x = (lo <= k0 && k0 <= lo + 8) ? 1.0f : 0.0f;
    bandA[s].y = (lo <= k0 + 1 && k0 + 1 <= lo + 8) ? 1.0f : 0.0f;
  }
#else
  v16h bandH;     // A operand, 16x32 f16: exact (weights are 1.0), K>=24 zeroed
#pragma unroll
  for (int v = 0; v < 8; ++v) {
#pragma unroll
    for (int t = 0; t < 2; ++t) {
      int k = (v < 4) ? (8 * hi + 2 * v + t) : (16 + 8 * hi + 2 * (v - 4) + t);
      bandH[2 * v + t] = (_Float16)((lo <= k && k <= lo + 8 && k < TIN) ? 1.0f : 0.0f);
    }
  }
#endif

  // running 9-slab sums, in WMMA C/D register layout: voxel (M=k+8*hi, N=lo)
  float rs[5][8];
#pragma unroll
  for (int q = 0; q < 5; ++q)
#pragma unroll
    for (int k = 0; k < 8; ++k) rs[q][k] = 0.0f;
  float acc = 0.0f;

  const int dbase = dz * DCH;
  const int dlo = dbase - HALO;
  const int dhi = dbase + DCH + HALO - 1;

  auto issue = [&](int d) {
    int dcl = min(max(d, 0), DDIM - 1);  // keep TENSORcnt cadence uniform at D edges
    size_t off = (((size_t)b * DDIM + dcl) * HDIM + r0) * WDIM + c0;
    int bb = d & 1;
    tdm_load_tile_2d(gI + off, lds_addr_of(&sI[bb][0]), cww, chh);
    tdm_load_tile_2d(gJ + off, lds_addr_of(&sJ[bb][0]), cww, chh);
  };

  issue(dlo);
  for (int d = dlo; d <= dhi; ++d) {
    if (d < dhi) {
      issue(d + 1);                           // prefetch next slab (other buffer)
      __builtin_amdgcn_s_wait_tensorcnt(2);   // current slab's 2 TDM ops done
    } else {
      __builtin_amdgcn_s_wait_tensorcnt(0);
    }
    const int buf = d & 1;
    const bool dvalid = (d >= 0) && (d < DDIM);

    v8f box[5];
    if (dvalid) {
      // ---- W-axis 9-tap sums (sliding window); lanes 0..23 own input rows ----
      if (lane < TIN) {
        const int rr = lane - lro;
        const bool rowok = (unsigned)rr < (unsigned)chh;
        const float* rowI = &sI[buf][rr * cww];
        const float* rowJ = &sJ[buf][rr * cww];
        float iv[TIN], jv[TIN];
#pragma unroll
        for (int c = 0; c < TIN; ++c) {
          int cc = c - lco;
          bool ok = rowok && ((unsigned)cc < (unsigned)cww);
          iv[c] = ok ? rowI[cc] : 0.0f;   // zero padding outside the volume
          jv[c] = ok ? rowJ[cc] : 0.0f;
        }
#pragma unroll
        for (int q = 0; q < 5; ++q) {     // 0:I 1:J 2:II 3:JJ 4:IJ
          float p[TIN];
#pragma unroll
          for (int c = 0; c < TIN; ++c) {
            float a = iv[c], bb2 = jv[c];
            p[c] = (q == 0) ? a : (q == 1) ? bb2 : (q == 2) ? a * a
                 : (q == 3) ? bb2 * bb2 : a * bb2;
          }
          float s = p[0];
#pragma unroll
          for (int k = 1; k < 9; ++k) s += p[k];
          float* wr = &sW[q][lane * PW];
          wr[0] = s;
#pragma unroll
          for (int c = 1; c < TS; ++c) { s += p[c + 8] - p[c - 1]; wr[c] = s; }
        }
      }

      // ---- stash center I/J (needed 4 slabs later for I_u, J_u) ----
      {
        const int ci = d % 5;
        float* dstI = &sCenI[ci][lane * 8];
        float* dstJ = &sCenJ[ci][lane * 8];
#pragma unroll
        for (int k = 0; k < 8; ++k) {
          int m = k + 8 * hi;
          int idx = (HALO + m - lro) * cww + (HALO + lo - lco);  // always in-clamp
          dstI[k] = sI[buf][idx];
          dstJ[k] = sJ[buf][idx];
        }
      }

      // ---- H-axis 9-tap sums on the matrix pipe: D = Band x Wsum ----
#pragma unroll
      for (int q = 0; q < 5; ++q) {
        v8f cacc = {};
#ifdef NCC_WMMA_F32X4
#pragma unroll
        for (int s = 0; s < 6; ++s) {
          int k0 = 4 * s + 2 * hi;
          v2f bop;                                   // B operand 4x16: lane -> N=lo
          bop.x = sW[q][k0 * PW + lo];
          bop.y = sW[q][(k0 + 1) * PW + lo];
          cacc = __builtin_amdgcn_wmma_f32_16x16x4_f32(
              false, bandA[s], false, bop, (short)0, cacc, false, false);
        }
#else
        v16h bop;                                    // B operand 32x16 f16
#pragma unroll
        for (int v = 0; v < 8; ++v) {
#pragma unroll
          for (int t = 0; t < 2; ++t) {
            int k = (v < 4) ? (8 * hi + 2 * v + t) : (16 + 8 * hi + 2 * (v - 4) + t);
            float x = (k < TIN) ? sW[q][k * PW + lo] : 0.0f;
            bop[2 * v + t] = (_Float16)x;
          }
        }
        cacc = __builtin_amdgcn_wmma_f32_16x16x32_f16(
            false, bandH, false, bop, (short)0, cacc, false, false);
#endif
        box[q] = cacc;
      }
    } else {
#pragma unroll
      for (int q = 0; q < 5; ++q) {
        v8f z = {};
        box[q] = z;                                  // zero slab outside the volume
      }
    }

    // ---- D-axis: ring update + running window sums (add new, subtract old) ----
    const int slot = ((d % 9) + 9) % 9;
#pragma unroll
    for (int q = 0; q < 5; ++q) {
      float* slab = &sRing[slot][q][lane * 8];
#pragma unroll
      for (int k = 0; k < 8; ++k) {
        float old = slab[k];
        float nw = box[q][k];
        slab[k] = nw;
        rs[q][k] += nw - old;
      }
    }

    // ---- emit NCC terms for center depth dc = d-4 ----
    const int dc = d - HALO;
    if (dc >= dbase && dc < dbase + DCH) {
      const int ci = dc % 5;
      const float* cIp = &sCenI[ci][lane * 8];
      const float* cJp = &sCenJ[ci][lane * 8];
#pragma unroll
      for (int k = 0; k < 8; ++k) {
        float Iv = cIp[k], Jv = cJp[k];
        float Iu = Iv * (1.0f / KSUMF);
        float Ju = Jv * (1.0f / KSUMF);
        float Is = rs[0][k], Js = rs[1][k];
        float IIs = rs[2][k], JJs = rs[3][k], IJs = rs[4][k];
        float cross = IJs - Is * Ju - Js * Iu + Iu * Ju * KSUMF;
        float Ivar = IIs - 2.0f * Is * Iu + Iu * Iu * KSUMF;
        float Jvar = JJs - 2.0f * Js * Ju + Ju * Ju * KSUMF;
        acc += (cross * cross) / (Ivar * Jvar + 1e-5f);
      }
    }
  }

  // wave32 reduction -> one partial per block (deterministic)
#pragma unroll
  for (int off2 = 16; off2 > 0; off2 >>= 1) acc += __shfl_xor(acc, off2, 32);
  if (lane == 0) partials[bid] = acc;
}

__global__ __launch_bounds__(32) void ncc_finalize(const float* __restrict__ partials,
                                                   float* __restrict__ out) {
  double s = 0.0;
  for (int i = threadIdx.x; i < NBLK; i += 32) s += (double)partials[i];
#pragma unroll
  for (int off = 16; off > 0; off >>= 1) s += __shfl_xor(s, off, 32);
  if (threadIdx.x == 0) out[0] = (float)(-s / NTOT);
}

extern "C" void kernel_launch(void* const* d_in, const int* in_sizes, int n_in,
                              void* d_out, int out_size, void* d_ws, size_t ws_size,
                              hipStream_t stream) {
  (void)in_sizes; (void)n_in; (void)out_size; (void)ws_size;
  const float* I = (const float*)d_in[0];
  const float* J = (const float*)d_in[1];
  float* partials = (float*)d_ws;                 // NBLK floats of scratch
  ncc_main<<<NBLK, 32, 0, stream>>>(I, J, partials);
  ncc_finalize<<<1, 32, 0, stream>>>(partials, (float*)d_out);
}
